// StructuredTransformer_13039520711128
// MI455X (gfx1250) — compile-verified, matrix operand order checked by reading, not run on hardware
//
#include <hip/hip_runtime.h>
#include <math.h>

#define N_NODES 8192
#define KNBR    24
#define NEDGE   (N_NODES * KNBR)          // 196608
#define DMODEL  128
#define HDIM    16
#define ENC_K   153
#define DEC_K   169
#define ENC_STRIDE 160                     // ceil(153/32)*32
#define DEC_STRIDE 192                     // ceil(169/32)*32
#define OUT_DIM 20
#define SEQ_DIM 16

typedef __attribute__((ext_vector_type(16))) __bf16 v16bf;
typedef __attribute__((ext_vector_type(8)))  __bf16 v8bf;
typedef __attribute__((ext_vector_type(4)))  __bf16 v4bf;
typedef __attribute__((ext_vector_type(8)))  float  v8f;

// ---------------------------------------------------------------------------
// LDS-free tiled WMMA GEMM:
//   C[M x Nc] = epilogue( A[M x KP] @ Wt^T + b ),  KP = KC*32 (compile time)
//  - A: fp32 (optionally pre-ReLU'd, row stride KP) or zero-padded bf16
//    (row stride KP).
//  - Wt: pre-packed bf16, transposed + padded: Wt[n * KP + k].
//  - 128 threads = 4 waves; wave owns 16 rows; block covers NT*16 columns
//    (grid.y splits columns). All operands loaded straight into the WMMA
//    VGPR layout (K = group*16 + half*8 + j) with 16-byte global loads.
//  - K loop fully unrolled: KC * NT v_wmma_f32_16x16x32_bf16 per wave.
// ---------------------------------------------------------------------------
template<int NT, int KC, bool ABF16, bool PRERELU, bool RELU, bool RESID>
__global__ __launch_bounds__(128)
void gemm_wmma(const void* __restrict__ Aptr,
               const __bf16* __restrict__ Wt,
               const float* __restrict__ bias, int Nc,
               const float* __restrict__ resid,
               float* __restrict__ C, int ldc)
{
  constexpr int KP = KC * 32;
  const int tid  = threadIdx.x;
  const int wave = tid >> 5;
  const int lane = tid & 31;
  const int hl   = lane >> 4;
  const int l16  = lane & 15;
  const long m0  = (long)blockIdx.x * 64;
  const int col0 = blockIdx.y * (NT * 16);
  const long mrow = m0 + wave * 16 + l16;

  const float*  Af = (const float*)Aptr;
  const __bf16* Ab = (const __bf16*)Aptr;

  v8f acc[NT];
#pragma unroll
  for (int c = 0; c < NT; ++c) {
#pragma unroll
    for (int i = 0; i < 8; ++i) acc[c][i] = 0.0f;
  }

#pragma unroll
  for (int kc = 0; kc < KC; ++kc) {
    const int k0 = kc * 32;

    // ---- A fragment straight into WMMA layout ----
    v16bf afrag;
    if (ABF16) {
      const __bf16* ar = Ab + mrow * (long)KP + k0 + hl * 8;
      v8bf alo = *(const v8bf*)ar;
      v8bf ahi = *(const v8bf*)(ar + 16);
#pragma unroll
      for (int i = 0; i < 8; ++i) { afrag[i] = alo[i]; afrag[8 + i] = ahi[i]; }
    } else {
      const float* ar = Af + mrow * (long)KP + k0 + hl * 8;
      float4 f0 = *(const float4*)(ar);
      float4 f1 = *(const float4*)(ar + 4);
      float4 f2 = *(const float4*)(ar + 16);
      float4 f3 = *(const float4*)(ar + 20);
      float v[16] = { f0.x, f0.y, f0.z, f0.w, f1.x, f1.y, f1.z, f1.w,
                      f2.x, f2.y, f2.z, f2.w, f3.x, f3.y, f3.z, f3.w };
#pragma unroll
      for (int i = 0; i < 16; ++i) {
        float x = PRERELU ? fmaxf(v[i], 0.0f) : v[i];
        afrag[i] = (__bf16)x;
      }
    }

    // ---- B fragments from pre-packed transposed weights + WMMA ----
    const __bf16* wb = Wt + (long)col0 * KP + k0 + hl * 8;
#pragma unroll
    for (int c = 0; c < NT; ++c) {
      const __bf16* br = wb + (long)(c * 16 + l16) * KP;
      v8bf blo = *(const v8bf*)br;
      v8bf bhi = *(const v8bf*)(br + 16);
      v16bf bfrag;
#pragma unroll
      for (int i = 0; i < 8; ++i) { bfrag[i] = blo[i]; bfrag[8 + i] = bhi[i]; }
      acc[c] = __builtin_amdgcn_wmma_f32_16x16x32_bf16(
          false, afrag, false, bfrag, (short)0, acc[c], false, false);
    }
  }

  // ---- epilogue: bias, residual, relu, store ----
#pragma unroll
  for (int c = 0; c < NT; ++c) {
    int cg = col0 + c * 16 + l16;
    if (cg < Nc) {
      float bb = bias[cg];
#pragma unroll
      for (int r = 0; r < 8; ++r) {
        long m = m0 + wave * 16 + r + 8 * hl;   // D-matrix VGPR layout
        float v = acc[c][r] + bb;
        if (RESID) v += resid[m * (long)ldc + cg];
        if (RELU)  v = fmaxf(v, 0.0f);
        C[m * (long)ldc + cg] = v;
      }
    }
  }
}

// Pack fp32 W[Kdim x Nc] -> bf16 Wt[Ncpad x Kpad] (transposed, zero padded).
__global__ void pack_weight(const float* __restrict__ W, int Kdim, int Nc,
                            int Kpad, __bf16* __restrict__ Wt)
{
  int n = blockIdx.x;                       // 0..Ncpad-1
  int k = blockIdx.y * 64 + threadIdx.x;
  if (k >= Kpad) return;
  float v = (k < Kdim && n < Nc) ? W[(long)k * Nc + n] : 0.0f;
  Wt[(long)n * Kpad + k] = (__bf16)v;
}

// ---------------------------------------------------------------------------
// Relative edge features: rbf(16) | direction(3) | quat(4) | sin | cos -> 25
// ---------------------------------------------------------------------------
__device__ __forceinline__ float sgnf(float v) {
  return (v > 0.0f) ? 1.0f : ((v < 0.0f) ? -1.0f : 0.0f);
}

__global__ void rel_kernel(const float* __restrict__ dist,
                           const int* __restrict__ conn,
                           const int* __restrict__ idx,
                           float* __restrict__ rel)
{
  int e = blockIdx.x * blockDim.x + threadIdx.x;
  if (e >= NEDGE) return;
  const float* src = dist + (long)conn[e] * 13;
  const float* tgt = dist + (long)idx[e]  * 13;

  float d0 = src[0] - tgt[0];
  float d1 = src[1] - tgt[1];
  float d2 = src[2] - tgt[2];
  float dn = sqrtf(d0 * d0 + d1 * d1 + d2 * d2);

  float xo[9], yo[9];
#pragma unroll
  for (int i = 0; i < 9; ++i) { xo[i] = tgt[3 + i]; yo[i] = src[3 + i]; }

  float invd = 1.0f / (dn + 1e-8f);
  float dir[3];
#pragma unroll
  for (int j = 0; j < 3; ++j)
    dir[j] = (xo[0 * 3 + j] * d0 + xo[1 * 3 + j] * d1 + xo[2 * 3 + j] * d2) * invd;

  float R[3][3];
#pragma unroll
  for (int i = 0; i < 3; ++i)
#pragma unroll
    for (int j = 0; j < 3; ++j)
      R[i][j] = xo[0 * 3 + i] * yo[0 * 3 + j] + xo[1 * 3 + i] * yo[1 * 3 + j] +
                xo[2 * 3 + i] * yo[2 * 3 + j];

  float Rxx = R[0][0], Ryy = R[1][1], Rzz = R[2][2];
  float m0 = 0.5f * sqrtf(fabsf(1.0f + Rxx - Ryy - Rzz));
  float m1 = 0.5f * sqrtf(fabsf(1.0f - Rxx + Ryy - Rzz));
  float m2 = 0.5f * sqrtf(fabsf(1.0f - Rxx - Ryy + Rzz));
  float m3 = 0.5f * sqrtf(fabsf(1.0f + Rxx + Ryy + Rzz));
  float q0 = m0 * sgnf(R[2][1] - R[1][2]);
  float q1 = m1 * sgnf(R[0][2] - R[2][0]);
  float q2 = m2 * sgnf(R[1][0] - R[0][1]);
  float q3 = m3;
  float qn = 1.0f / (sqrtf(q0 * q0 + q1 * q1 + q2 * q2 + q3 * q3) + 1e-8f);

  float* out = rel + (long)e * 25;
  const float sigma = 20.0f / 15.0f;
  const float isig  = 1.0f / sigma;
#pragma unroll
  for (int k = 0; k < 16; ++k) {
    float t = (dn - (float)k * sigma) * isig;
    out[k] = expf(-t * t);
  }
  out[16] = dir[0]; out[17] = dir[1]; out[18] = dir[2];
  out[19] = q0 * qn; out[20] = q1 * qn; out[21] = q2 * qn; out[22] = q3 * qn;
  float phase = (tgt[12] - src[12]) * 0.1f;
  out[23] = sinf(phase);
  out[24] = cosf(phase);
}

// seq_emb[n] = seq_w[sequence[n]] + seq_b   (one-hot matmul == row gather)
__global__ void seq_kernel(const int* __restrict__ seqids,
                           const float* __restrict__ w,
                           const float* __restrict__ b,
                           float* __restrict__ out)
{
  int i = blockIdx.x * blockDim.x + threadIdx.x;
  if (i >= N_NODES * SEQ_DIM) return;
  int n = i >> 4, j = i & 15;
  out[i] = w[(long)seqids[n] * SEQ_DIM + j] + b[j];
}

// ---------------------------------------------------------------------------
// Edge-matrix builders: vectorized float4 -> 4x bf16, no div/mod.
// Main section: columns 0..127 (gathered node features).
// ---------------------------------------------------------------------------
template<bool DEC>
__global__ void build_edge_main(__bf16* __restrict__ A, int stride,
                                const float* __restrict__ local,
                                const float* __restrict__ enc,
                                const int* __restrict__ conn)
{
  long i = (long)blockIdx.x * 256 + threadIdx.x;   // over NEDGE*32
  if (i >= (long)NEDGE * 32) return;
  long e  = i >> 5;
  int  c4 = (int)(i & 31) << 2;
  int  cn = conn[e];
  const float* src = local;
  if (DEC) { if (cn >= (int)e) src = enc; }        // pre ? local : encoding
  float4 f = *(const float4*)&src[(long)cn * DMODEL + c4];
  v4bf b;
  b[0] = (__bf16)f.x; b[1] = (__bf16)f.y; b[2] = (__bf16)f.z; b[3] = (__bf16)f.w;
  *(v4bf*)&A[e * (long)stride + c4] = b;
}

// Encoder tail: columns 128..159 = rel(25) | zero pad(7)
__global__ void build_edge_tail_enc(__bf16* __restrict__ A,
                                    const float* __restrict__ rel)
{
  long i = (long)blockIdx.x * 256 + threadIdx.x;   // over NEDGE*32
  if (i >= (long)NEDGE * 32) return;
  long e = i >> 5;
  int  c = (int)(i & 31);
  float v = (c < 25) ? rel[e * 25 + c] : 0.0f;
  A[e * (long)ENC_STRIDE + 128 + c] = (__bf16)v;
}

// Decoder tail: columns 128..191 = rel(25) | gated seq_emb(16) | pad(23)
__global__ void build_edge_tail_dec(__bf16* __restrict__ A,
                                    const float* __restrict__ rel,
                                    const float* __restrict__ seqe,
                                    const int* __restrict__ conn)
{
  long i = (long)blockIdx.x * 256 + threadIdx.x;   // over NEDGE*64
  if (i >= (long)NEDGE * 64) return;
  long e = i >> 6;
  int  c = (int)(i & 63);
  float v = 0.0f;
  if (c < 25) {
    v = rel[e * 25 + c];
  } else if (c < 41) {
    int cn = conn[e];
    if (cn < (int)e) v = seqe[(long)cn * SEQ_DIM + (c - 25)];
  }
  A[e * (long)DEC_STRIDE + 128 + c] = (__bf16)v;
}

// ---------------------------------------------------------------------------
// Attention core: one block per node, thread = (head, dim).
// ---------------------------------------------------------------------------
__global__ __launch_bounds__(128)
void attn_kernel(const float* __restrict__ q,
                 const float* __restrict__ kmat,
                 const float* __restrict__ vmat,
                 float* __restrict__ o)
{
  int n = blockIdx.x;
  int t = threadIdx.x;
  int h = t >> 4, d = t & 15;

  float qv = q[(long)n * DMODEL + h * HDIM + d] * 0.25f;   // 1/sqrt(16)
  const float* kb = kmat + (long)n * KNBR * DMODEL + h * HDIM + d;
  const float* vb = vmat + (long)n * KNBR * DMODEL + h * HDIM + d;

  float lg[KNBR];
#pragma unroll
  for (int k = 0; k < KNBR; ++k) {
    float p = qv * kb[(long)k * DMODEL];
    p += __shfl_xor(p, 1, 16);
    p += __shfl_xor(p, 2, 16);
    p += __shfl_xor(p, 4, 16);
    p += __shfl_xor(p, 8, 16);
    lg[k] = p;
  }
  float mx = lg[0];
#pragma unroll
  for (int k = 1; k < KNBR; ++k) mx = fmaxf(mx, lg[k]);
  float s = 0.0f;
#pragma unroll
  for (int k = 0; k < KNBR; ++k) { lg[k] = __expf(lg[k] - mx); s += lg[k]; }
  float inv = 1.0f / s;
  float acc = 0.0f;
#pragma unroll
  for (int k = 0; k < KNBR; ++k) acc += lg[k] * vb[(long)k * DMODEL];
  o[(long)n * DMODEL + t] = acc * inv;
}

// ---------------------------------------------------------------------------
// Host orchestration
// ---------------------------------------------------------------------------
struct LinP   { const float *b, *w; };
struct BlockP { LinP k, o, q, v, m0, m1, m2; };

#define GEMM(NT, KC, ABF16, PRE, RELU, RESID, M, A, Wt, Bp, Nc, Rp, Cp, ldc)       \
  gemm_wmma<NT, KC, ABF16, PRE, RELU, RESID>                                       \
      <<<dim3((M) / 64, ((Nc) + (NT)*16 - 1) / ((NT)*16)), 128, 0, stream>>>(      \
          A, Wt, Bp, Nc, Rp, Cp, ldc)

extern "C" void kernel_launch(void* const* d_in, const int* in_sizes, int n_in,
                              void* d_out, int out_size, void* d_ws, size_t ws_size,
                              hipStream_t stream)
{
  (void)in_sizes; (void)n_in; (void)out_size; (void)ws_size;

  const float* features    = (const float*)d_in[0];
  const int*   sequence    = (const int*)d_in[1];
  const float* distances   = (const float*)d_in[2];
  const int*   connections = (const int*)d_in[3];
  const int*   indices     = (const int*)d_in[4];

  // Param leaves in jax pytree (sorted-key) order:
  // dec[0..2]{att{k,o,q,v}{b,w}, mlp[0..2]{b,w}}, enc[0..2]{...},
  // post{b,w}, pre{b,w}, seq{b,w}
  int p = 5;
  auto takeLin = [&]() {
    LinP l{ (const float*)d_in[p], (const float*)d_in[p + 1] };
    p += 2;
    return l;
  };
  BlockP dec[3], enc[3];
  for (int i = 0; i < 3; ++i) {
    BlockP b;
    b.k = takeLin(); b.o = takeLin(); b.q = takeLin(); b.v = takeLin();
    b.m0 = takeLin(); b.m1 = takeLin(); b.m2 = takeLin();
    dec[i] = b;
  }
  for (int i = 0; i < 3; ++i) {
    BlockP b;
    b.k = takeLin(); b.o = takeLin(); b.q = takeLin(); b.v = takeLin();
    b.m0 = takeLin(); b.m1 = takeLin(); b.m2 = takeLin();
    enc[i] = b;
  }
  LinP postL = takeLin();
  LinP preL  = takeLin();
  LinP seqL  = takeLin();

  // ---- workspace carve-up ----
  char* ws = (char*)d_ws;
  size_t off = 0;
  auto alloc = [&](size_t bytes) {
    void* ptr = ws + off;
    off = (off + bytes + 255) & ~(size_t)255;
    return ptr;
  };
  float* rel    = (float*)alloc((size_t)NEDGE * 25 * 4);
  float* seqe   = (float*)alloc((size_t)N_NODES * SEQ_DIM * 4);
  float* s0     = (float*)alloc((size_t)N_NODES * DMODEL * 4);
  float* s1     = (float*)alloc((size_t)N_NODES * DMODEL * 4);
  float* s2     = (float*)alloc((size_t)N_NODES * DMODEL * 4);
  float* h1     = (float*)alloc((size_t)N_NODES * DMODEL * 4);
  float* h2     = (float*)alloc((size_t)N_NODES * DMODEL * 4);
  float* loc    = (float*)alloc((size_t)N_NODES * DMODEL * 4);
  float* qbuf   = (float*)alloc((size_t)N_NODES * DMODEL * 4);
  float* oatt   = (float*)alloc((size_t)N_NODES * DMODEL * 4);
  __bf16* Aedge = (__bf16*)alloc((size_t)NEDGE * DEC_STRIDE * 2);
  float* kmat   = (float*)alloc((size_t)NEDGE * DMODEL * 4);
  float* vmat   = (float*)alloc((size_t)NEDGE * DMODEL * 4);

  // ---- pack all weights to transposed/padded bf16 once ----
  auto packW = [&](const float* W, int Kdim, int Nc, int Kpad, int Ncpad) {
    __bf16* dst = (__bf16*)alloc((size_t)Kpad * Ncpad * 2);
    pack_weight<<<dim3(Ncpad, (Kpad + 63) / 64), 64, 0, stream>>>(W, Kdim, Nc, Kpad, dst);
    return (const __bf16*)dst;
  };
  struct BlockW { const __bf16 *k, *o, *q, *v, *m0, *m1, *m2; };
  BlockW encW[3], decW[3];
  for (int i = 0; i < 3; ++i) {
    encW[i].m0 = packW(enc[i].m0.w, 128, 128, 128, 128);
    encW[i].m1 = packW(enc[i].m1.w, 128, 128, 128, 128);
    encW[i].m2 = packW(enc[i].m2.w, 128, 128, 128, 128);
    encW[i].q  = packW(enc[i].q.w, 128, 128, 128, 128);
    encW[i].o  = packW(enc[i].o.w, 128, 128, 128, 128);
    encW[i].k  = packW(enc[i].k.w, ENC_K, 128, ENC_STRIDE, 128);
    encW[i].v  = packW(enc[i].v.w, ENC_K, 128, ENC_STRIDE, 128);
    decW[i].m0 = packW(dec[i].m0.w, 128, 128, 128, 128);
    decW[i].m1 = packW(dec[i].m1.w, 128, 128, 128, 128);
    decW[i].m2 = packW(dec[i].m2.w, 128, 128, 128, 128);
    decW[i].q  = packW(dec[i].q.w, 128, 128, 128, 128);
    decW[i].o  = packW(dec[i].o.w, 128, 128, 128, 128);
    decW[i].k  = packW(dec[i].k.w, DEC_K, 128, DEC_STRIDE, 128);
    decW[i].v  = packW(dec[i].v.w, DEC_K, 128, DEC_STRIDE, 128);
  }
  const __bf16* preW  = packW(preL.w,  64, 128, 64, 128);
  const __bf16* postW = packW(postL.w, 128, OUT_DIM, 128, 32);

  // ---- precompute edge geometry + sequence embedding ----
  rel_kernel<<<(NEDGE + 255) / 256, 256, 0, stream>>>(distances, connections, indices, rel);
  seq_kernel<<<(N_NODES * SEQ_DIM + 255) / 256, 256, 0, stream>>>(sequence, seqL.w, seqL.b, seqe);

  // ---- pre linear: features (N x 64) -> s0 (N x 128) ----
  GEMM(4, 2, false, false, false, false, N_NODES, features, preW,
       preL.b, DMODEL, nullptr, s0, DMODEL);

  float* x = s0;
  float* ping[2] = { s0, s1 };
  int cur = 0;

  const int EG32 = (int)(((long)NEDGE * 32 + 255) / 256);
  const int EG64 = (int)(((long)NEDGE * 64 + 255) / 256);

  // ---- encoder blocks ----
  for (int i = 0; i < 3; ++i) {
    const BlockP& bp = enc[i];
    const BlockW& bw = encW[i];
    GEMM(4, 4, false, true,  true,  false, N_NODES, x,  bw.m0, bp.m0.b, DMODEL, nullptr, h1,  DMODEL);
    GEMM(4, 4, false, false, true,  false, N_NODES, h1, bw.m1, bp.m1.b, DMODEL, nullptr, h2,  DMODEL);
    GEMM(4, 4, false, false, true,  false, N_NODES, h2, bw.m2, bp.m2.b, DMODEL, nullptr, loc, DMODEL);

    build_edge_main<false><<<EG32, 256, 0, stream>>>(Aedge, ENC_STRIDE, loc, nullptr, connections);
    build_edge_tail_enc<<<EG32, 256, 0, stream>>>(Aedge, rel);

    GEMM(4, 4, false, false, false, false, N_NODES, loc, bw.q, bp.q.b, DMODEL, nullptr, qbuf, DMODEL);
    GEMM(4, 5, true,  false, false, false, NEDGE, Aedge, bw.k, bp.k.b, DMODEL, nullptr, kmat, DMODEL);
    GEMM(4, 5, true,  false, false, false, NEDGE, Aedge, bw.v, bp.v.b, DMODEL, nullptr, vmat, DMODEL);

    attn_kernel<<<N_NODES, 128, 0, stream>>>(qbuf, kmat, vmat, oatt);

    float* xn = ping[cur ^ 1];
    GEMM(4, 4, false, false, false, true, N_NODES, oatt, bw.o, bp.o.b, DMODEL, x, xn, DMODEL);
    x = xn; cur ^= 1;
  }

  float* encoding = x;                 // pre->s0, b0->s1, b1->s0, b2->s1
  float* dping[2] = { (encoding == s0) ? s1 : s0, s2 };
  int dcur = 0;

  // ---- decoder blocks ----
  for (int i = 0; i < 3; ++i) {
    const BlockP& bp = dec[i];
    const BlockW& bw = decW[i];
    GEMM(4, 4, false, true,  true,  false, N_NODES, x,  bw.m0, bp.m0.b, DMODEL, nullptr, h1,  DMODEL);
    GEMM(4, 4, false, false, true,  false, N_NODES, h1, bw.m1, bp.m1.b, DMODEL, nullptr, h2,  DMODEL);
    GEMM(4, 4, false, false, true,  false, N_NODES, h2, bw.m2, bp.m2.b, DMODEL, nullptr, loc, DMODEL);

    build_edge_main<true><<<EG32, 256, 0, stream>>>(Aedge, DEC_STRIDE, loc, encoding, connections);
    build_edge_tail_dec<<<EG64, 256, 0, stream>>>(Aedge, rel, seqe, connections);

    GEMM(4, 4, false, false, false, false, N_NODES, loc, bw.q, bp.q.b, DMODEL, nullptr, qbuf, DMODEL);
    GEMM(4, 6, true,  false, false, false, NEDGE, Aedge, bw.k, bp.k.b, DMODEL, nullptr, kmat, DMODEL);
    GEMM(4, 6, true,  false, false, false, NEDGE, Aedge, bw.v, bp.v.b, DMODEL, nullptr, vmat, DMODEL);

    attn_kernel<<<N_NODES, 128, 0, stream>>>(qbuf, kmat, vmat, oatt);

    float* xn = dping[dcur];
    GEMM(4, 4, false, false, false, true, N_NODES, oatt, bw.o, bp.o.b, DMODEL, x, xn, DMODEL);
    x = xn; dcur ^= 1;
  }

  // ---- post linear: x (N x 128) -> d_out (N x 20) ----
  GEMM(2, 4, false, false, false, false, N_NODES, x, postW,
       postL.b, OUT_DIM, nullptr, (float*)d_out, OUT_DIM);
}
